// HierarchicalProteinGNN_1812476199678
// MI455X (gfx1250) — matrix operand order repaired and checked by reading, not compiled, not run on hardware
//
#include <hip/hip_runtime.h>
#include <hip/hip_bf16.h>
#include <math.h>

// ---------------- problem constants (match reference) ----------------
#define NRES  20000
#define FIN   1302
#define KP1   1312      // FIN padded to 32
#define HD    512       // H*D
#define NH    4
#define DH    128
#define NSSE  2000
#define SSEIN 132
#define KPS   160       // SSEIN padded to 32
#define NDOM  40
#define ERES  300000
#define ESSE  8000
#define EDOM  160
#define OUTC  256

#define CDIV(a,b) (((a)+(b)-1)/(b))

typedef __bf16 bf16;
typedef __attribute__((ext_vector_type(16))) __bf16 v16bf;
typedef __attribute__((ext_vector_type(8)))  __bf16 bf16x8;
typedef __attribute__((ext_vector_type(8)))  float  v8f;

// ---------------- helpers ----------------
__device__ __forceinline__ float lrelu(float x) { return x > 0.f ? x : 0.2f * x; }

// order-preserving float<->uint encoding for atomic max on f32
__device__ __forceinline__ unsigned fenc(float f) {
  unsigned u = __float_as_uint(f);
  return (u & 0x80000000u) ? ~u : (u | 0x80000000u);
}
__device__ __forceinline__ float fdec(unsigned u) {
  return (u & 0x80000000u) ? __uint_as_float(u & 0x7FFFFFFFu) : __uint_as_float(~u);
}
#define ENC_NEG_INF 0x007FFFFFu   // fenc(-inf)

// ---------------- conversion / fill kernels ----------------
__global__ void cvt_pad(const float* __restrict__ in, bf16* __restrict__ out,
                        int M, int K, int Kp) {
  long long gid = (long long)blockIdx.x * blockDim.x + threadIdx.x;
  if (gid >= (long long)M * Kp) return;
  int r = (int)(gid / Kp), k = (int)(gid % Kp);
  out[gid] = (k < K) ? (bf16)in[(long long)r * K + k] : (bf16)0.0f;
}

__global__ void cvt_pad_t(const float* __restrict__ W, bf16* __restrict__ out,
                          int K, int N, int Kp) {
  long long gid = (long long)blockIdx.x * blockDim.x + threadIdx.x;
  if (gid >= (long long)N * Kp) return;
  int n = (int)(gid / Kp), k = (int)(gid % Kp);
  out[gid] = (k < K) ? (bf16)W[(long long)k * N + n] : (bf16)0.0f;
}

__global__ void fill_f32(float* p, float v, long long n) {
  long long gid = (long long)blockIdx.x * blockDim.x + threadIdx.x;
  if (gid < n) p[gid] = v;
}
__global__ void fill_u32(unsigned* p, unsigned v, long long n) {
  long long gid = (long long)blockIdx.x * blockDim.x + threadIdx.x;
  if (gid < n) p[gid] = v;
}
__global__ void fill_bias(float* out, const float* __restrict__ bias, int N, int C) {
  long long gid = (long long)blockIdx.x * blockDim.x + threadIdx.x;
  if (gid >= (long long)N * C) return;
  out[gid] = bias[gid % C];
}

// ---------------- WMMA GEMM:  C[MxN] = A[MxKp](bf16) * Bt[NxKp](bf16)^T ----------------
// A row-major K-contiguous, B stored transposed (N rows of Kp) so both operand
// fragments are contiguous 16B chunks -> global_load_b128 feeding v_wmma.
// Each wave computes a 16x64 output strip (4 N-tiles) reusing one A fragment
// across 4 WMMAs: 10 b128 loads per 4 wmma instead of 16.
#define NT 4   // 16-wide N tiles per wave
__global__ void __launch_bounds__(256) wmma_gemm_tn(
    const bf16* __restrict__ A, const bf16* __restrict__ Bt,
    float* __restrict__ C, int M, int N, int Kp) {
  const int lane = threadIdx.x & 31;
  const int wave = threadIdx.x >> 5;
  const int half = lane >> 4;      // 0: lanes 0-15, 1: lanes 16-31
  const int l16  = lane & 15;
  const int m0 = (blockIdx.y * 8 + wave) * 16;
  const int n0 = blockIdx.x * (16 * NT);
  if (m0 >= M) return;             // wave-uniform: EXEC stays all-ones for WMMA

  // A fragment rows: M = l16 for both lane halves; K chunks {0..7|8..15} and {16..23|24..31}
  const bf16* arow = A  + (size_t)(m0 + l16) * Kp + half * 8;
  // B fragment: column N = n0 + j*16 + l16; K chunks {0..7|8..15} and {16..23|24..31} (same as A)
  const bf16* brow = Bt + (size_t)(n0 + l16) * Kp + half * 8;
  const size_t bstride = (size_t)16 * Kp;   // stride between adjacent N-tiles

  v8f acc[NT];
#pragma unroll
  for (int j = 0; j < NT; ++j) acc[j] = (v8f){};

  union U { v16bf v; bf16x8 h[2]; };
  for (int k0 = 0; k0 < Kp; k0 += 32) {
    U a;
    a.h[0] = *(const bf16x8*)(arow + k0);          // K k0+{0..7} (+8 for half=1)
    a.h[1] = *(const bf16x8*)(arow + k0 + 16);     // K k0+{16..23} (+8 for half=1)
#pragma unroll
    for (int j = 0; j < NT; ++j) {
      U b;
      b.h[0] = *(const bf16x8*)(brow + j * bstride + k0);      // K k0+half*8+{0..7}
      b.h[1] = *(const bf16x8*)(brow + j * bstride + k0 + 16); //  k0+16+half*8+{0..7}
      acc[j] = __builtin_amdgcn_wmma_f32_16x16x32_bf16(
          /*neg_a=*/false, a.v, /*neg_b=*/false, b.v,
          /*c_mod=*/(short)0, acc[j], /*reuse_a=*/false, /*reuse_b=*/false);
    }
  }
  // C/D layout: VGPR r -> row m0 + half*8 + r, col n0 + j*16 + l16
  float* crow = C + (size_t)(m0 + half * 8) * N + n0 + l16;
#pragma unroll
  for (int j = 0; j < NT; ++j)
#pragma unroll
    for (int r = 0; r < 8; ++r) crow[(size_t)r * N + j * 16] = acc[j][r];
}

// ---------------- GAT attention ----------------
// es[n,h] = <h[n,h,:], a_src[h,:]>, ed likewise
__global__ void attn_scores(const float* __restrict__ Hm,
                            const float* __restrict__ asrc,
                            const float* __restrict__ adst,
                            float* __restrict__ es, float* __restrict__ ed, int N) {
  long long gid = (long long)blockIdx.x * blockDim.x + threadIdx.x;
  if (gid >= (long long)N * NH) return;
  int n = (int)(gid / NH), h = (int)(gid % NH);
  const float* hp = Hm + (size_t)n * HD + h * DH;
  const float* as = asrc + h * DH;
  const float* ad = adst + h * DH;
  float s = 0.f, d = 0.f;
  for (int k = 0; k < DH; ++k) { float x = hp[k]; s += x * as[k]; d += x * ad[k]; }
  es[gid] = s; ed[gid] = d;
}

__global__ void edge_max(const int* __restrict__ src, const int* __restrict__ dst,
                         int E, int N, const float* __restrict__ es,
                         const float* __restrict__ ed, unsigned* smax) {
  long long gid = (long long)blockIdx.x * blockDim.x + threadIdx.x;
  if (gid >= (long long)(E + N) * NH) return;
  int eid = (int)(gid >> 2), h = (int)(gid & 3);
  int s = eid < E ? src[eid] : eid - E;
  int d = eid < E ? dst[eid] : eid - E;
  float e = lrelu(es[s * NH + h] + ed[d * NH + h]);
  atomicMax(&smax[d * NH + h], fenc(e));
}

__global__ void edge_den(const int* __restrict__ src, const int* __restrict__ dst,
                         int E, int N, const float* __restrict__ es,
                         const float* __restrict__ ed,
                         const unsigned* __restrict__ smax, float* den) {
  long long gid = (long long)blockIdx.x * blockDim.x + threadIdx.x;
  if (gid >= (long long)(E + N) * NH) return;
  int eid = (int)(gid >> 2), h = (int)(gid & 3);
  int s = eid < E ? src[eid] : eid - E;
  int d = eid < E ? dst[eid] : eid - E;
  float e = lrelu(es[s * NH + h] + ed[d * NH + h]);
  atomicAdd(&den[d * NH + h], expf(e - fdec(smax[d * NH + h])));
}

// one block (128 threads = D channels) per (edge,head); scatter-add alpha*h[src]
__global__ void __launch_bounds__(128) edge_acc(
    const int* __restrict__ src, const int* __restrict__ dst, int E, int N,
    const float* __restrict__ es, const float* __restrict__ ed,
    const unsigned* __restrict__ smax, const float* __restrict__ den,
    const float* __restrict__ Hm, float* ob) {
  int bid = blockIdx.x;
  int eid = bid >> 2, h = bid & 3;
  int s = eid < E ? src[eid] : eid - E;
  int d = eid < E ? dst[eid] : eid - E;
  float e  = lrelu(es[s * NH + h] + ed[d * NH + h]);
  float mx = fdec(smax[d * NH + h]);
  float alpha = expf(e - mx) / (den[d * NH + h] + 1e-16f);
  int dd = threadIdx.x;
  atomicAdd(&ob[(size_t)d * HD + h * DH + dd],
            alpha * Hm[(size_t)s * HD + h * DH + dd]);
}

// ---------------- BatchNorm (training stats) + ReLU ----------------
__global__ void __launch_bounds__(256) bn_stats(const float* __restrict__ X,
                                                int N, int C, float* mean, float* var) {
  int c = blockIdx.x, t = threadIdx.x;
  float s = 0.f, q = 0.f;
  for (int r = t; r < N; r += 256) {
    float x = X[(size_t)r * C + c]; s += x; q += x * x;
  }
  __shared__ float ss[256], qq[256];
  ss[t] = s; qq[t] = q; __syncthreads();
  for (int k = 128; k > 0; k >>= 1) {
    if (t < k) { ss[t] += ss[t + k]; qq[t] += qq[t + k]; }
    __syncthreads();
  }
  if (t == 0) { float m = ss[0] / N; mean[c] = m; var[c] = qq[0] / N - m * m; }
}

__global__ void bn_relu(float* X, int N, int C,
                        const float* __restrict__ mean, const float* __restrict__ var,
                        const float* __restrict__ g, const float* __restrict__ b) {
  long long gid = (long long)blockIdx.x * blockDim.x + threadIdx.x;
  if (gid >= (long long)N * C) return;
  int c = (int)(gid % C);
  float y = (X[gid] - mean[c]) * rsqrtf(var[c] + 1e-5f) * g[c] + b[c];
  X[gid] = y > 0.f ? y : 0.f;
}

// ---------------- head mean (concat=False) + bias ----------------
__global__ void head_mean(const float* __restrict__ ob, const float* __restrict__ bias,
                          float* xo, int N) {
  long long gid = (long long)blockIdx.x * blockDim.x + threadIdx.x;
  if (gid >= (long long)N * DH) return;
  int n = (int)(gid / DH), dd = (int)(gid % DH);
  float s = 0.f;
  for (int h = 0; h < NH; ++h) s += ob[(size_t)n * HD + h * DH + dd];
  xo[gid] = s * 0.25f + bias[dd];
}

// ---------------- pooling ----------------
__global__ void pool_cnt(const int* __restrict__ sid, float* cnt, int N) {
  int gid = blockIdx.x * blockDim.x + threadIdx.x;
  if (gid < N) atomicAdd(&cnt[sid[gid]], 1.0f);
}
__global__ void pool_add(const float* __restrict__ x, const int* __restrict__ sid,
                         float* pooled, int N, int C) {
  long long gid = (long long)blockIdx.x * blockDim.x + threadIdx.x;
  if (gid >= (long long)N * C) return;
  int n = (int)(gid / C), c = (int)(gid % C);
  atomicAdd(&pooled[(size_t)sid[n] * C + c], x[gid]);
}
__global__ void build_sse(const float* __restrict__ pooled, const float* __restrict__ cnt,
                          const int* __restrict__ sstype, float* xsse) {
  long long gid = (long long)blockIdx.x * blockDim.x + threadIdx.x;
  if (gid >= (long long)NSSE * SSEIN) return;
  int n = (int)(gid / SSEIN), c = (int)(gid % SSEIN);
  float cn = cnt[n];
  float v;
  if (c < DH)            v = pooled[(size_t)n * DH + c] / fmaxf(cn, 1.0f);
  else if (c < DH + 3)   v = (sstype[n] == (c - DH)) ? 1.0f : 0.0f;
  else                   v = cn / (float)NRES;
  xsse[gid] = v;
}
__global__ void dom_cnt(float* dcnt) {
  int gid = blockIdx.x * blockDim.x + threadIdx.x;
  if (gid < NSSE) atomicAdd(&dcnt[min(gid / (NSSE / NDOM), NDOM - 1)], 1.0f);
}
__global__ void dom_pool(const float* __restrict__ xs, float* xdom) {
  long long gid = (long long)blockIdx.x * blockDim.x + threadIdx.x;
  if (gid >= (long long)NSSE * DH) return;
  int n = (int)(gid / DH), c = (int)(gid % DH);
  int d = min(n / (NSSE / NDOM), NDOM - 1);
  atomicAdd(&xdom[(size_t)d * DH + c], xs[gid]);
}
__global__ void divide_rows(float* X, const float* __restrict__ cnt, int N, int C) {
  long long gid = (long long)blockIdx.x * blockDim.x + threadIdx.x;
  if (gid >= (long long)N * C) return;
  X[gid] /= fmaxf(cnt[gid / C], 1.0f);
}

// ---------------- domain GCN ----------------
__global__ void deg_count(const int* __restrict__ src, const int* __restrict__ dst,
                          int E, int N, float* deg) {
  int gid = blockIdx.x * blockDim.x + threadIdx.x;
  if (gid >= E + N) return;
  int d = gid < E ? dst[gid] : gid - E;
  atomicAdd(&deg[d], 1.0f);
}
__global__ void small_gemm(const float* __restrict__ X, const float* __restrict__ W,
                           float* Y, int K, int Ncol) {
  int r = blockIdx.x, c = threadIdx.x;
  float v = 0.f;
  for (int k = 0; k < K; ++k) v += X[r * K + k] * W[k * Ncol + c];
  Y[r * Ncol + c] = v;
}
__global__ void gcn_acc(const int* __restrict__ src, const int* __restrict__ dst,
                        int E, int N, const float* __restrict__ deg,
                        const float* __restrict__ Hm, float* ob, int C) {
  int e = blockIdx.x, c = threadIdx.x;
  int s = e < E ? src[e] : e - E;
  int d = e < E ? dst[e] : e - E;
  float coef = rsqrtf(fmaxf(deg[s], 1e-12f)) * rsqrtf(fmaxf(deg[d], 1e-12f));
  atomicAdd(&ob[(size_t)d * C + c], coef * Hm[(size_t)s * C + c]);
}

// ---------------- readout ----------------
__global__ void __launch_bounds__(256) colmean(const float* __restrict__ X,
                                               int N, int C, float* out) {
  int c = blockIdx.x, t = threadIdx.x;
  float s = 0.f;
  for (int r = t; r < N; r += 256) s += X[(size_t)r * C + c];
  __shared__ float ss[256];
  ss[t] = s; __syncthreads();
  for (int k = 128; k > 0; k >>= 1) { if (t < k) ss[t] += ss[t + k]; __syncthreads(); }
  if (t == 0) out[c] = ss[0] / N;
}
__global__ void proj_cat(const float* __restrict__ sseg, const float* __restrict__ Wp,
                         const float* __restrict__ bp, const float* __restrict__ dmean,
                         float* cat) {
  int j = threadIdx.x;  // 256
  float v = bp[j];
  for (int i = 0; i < DH; ++i) v += sseg[i] * Wp[i * OUTC + j];
  cat[j] = v;
  cat[OUTC + j] = dmean[j];
}
__global__ void __launch_bounds__(256) final_out(const float* __restrict__ cat,
                                                 const float* __restrict__ Wr,
                                                 const float* __restrict__ br,
                                                 float* out) {
  int j = threadIdx.x;
  float v = br[j];
  for (int i = 0; i < 2 * OUTC; ++i) v += cat[i] * Wr[i * OUTC + j];
  __shared__ float red[256];
  red[j] = v * v; __syncthreads();
  for (int k = 128; k > 0; k >>= 1) { if (j < k) red[j] += red[j + k]; __syncthreads(); }
  out[j] = v / fmaxf(sqrtf(red[0]), 1e-12f);
}

// ---------------- host orchestration ----------------
extern "C" void kernel_launch(void* const* d_in, const int* in_sizes, int n_in,
                              void* d_out, int out_size, void* d_ws, size_t ws_size,
                              hipStream_t stream) {
  // inputs in setup_inputs() dict order
  const float* x_res   = (const float*)d_in[0];
  const int*   ei_res  = (const int*)d_in[1];
  const int*   ei_sse  = (const int*)d_in[2];
  const int*   ei_dom  = (const int*)d_in[3];
  const int*   res_sid = (const int*)d_in[4];
  const int*   ss_type = (const int*)d_in[5];
  // d_in[6] = n_dom (compile-time NDOM)
  const float* W1 = (const float*)d_in[7];
  const float* a_src1 = (const float*)d_in[8];
  const float* a_dst1 = (const float*)d_in[9];
  const float* b1 = (const float*)d_in[10];
  const float* g1 = (const float*)d_in[11];
  const float* be1 = (const float*)d_in[12];
  const float* W2 = (const float*)d_in[13];
  const float* a_src2 = (const float*)d_in[14];
  const float* a_dst2 = (const float*)d_in[15];
  const float* b2 = (const float*)d_in[16];
  const float* g2 = (const float*)d_in[17];
  const float* be2 = (const float*)d_in[18];
  const float* Ws = (const float*)d_in[19];
  const float* a_srcs = (const float*)d_in[20];
  const float* a_dsts = (const float*)d_in[21];
  const float* bs = (const float*)d_in[22];
  const float* gs = (const float*)d_in[23];
  const float* bes = (const float*)d_in[24];
  const float* Wd = (const float*)d_in[25];
  const float* bd = (const float*)d_in[26];
  const float* gd = (const float*)d_in[27];
  const float* bed = (const float*)d_in[28];
  const float* Wproj = (const float*)d_in[29];
  const float* bproj = (const float*)d_in[30];
  const float* Wread = (const float*)d_in[31];
  const float* bread = (const float*)d_in[32];

  const int* src_r = ei_res;           const int* dst_r = ei_res + ERES;
  const int* src_s = ei_sse;           const int* dst_s = ei_sse + ESSE;
  const int* src_d = ei_dom;           const int* dst_d = ei_dom + EDOM;

  // workspace carve-out (256B aligned slabs)
  char* w = (char*)d_ws; size_t off = 0;
  auto carve = [&](size_t bytes) -> void* {
    void* p = w + off; off = (off + bytes + 255) & ~(size_t)255; return p;
  };
  bf16*     ABF   = (bf16*)carve((size_t)NRES * KP1 * 2);     // gemm A staging (reused)
  bf16*     WTBF  = (bf16*)carve((size_t)HD * KP1 * 2);       // transposed weights (reused)
  float*    HB    = (float*)carve((size_t)NRES * HD * 4);     // gemm output / GAT h
  float*    OB    = (float*)carve((size_t)NRES * HD * 4);     // GAT accumulation
  float*    ES    = (float*)carve((size_t)NRES * NH * 4);
  float*    ED    = (float*)carve((size_t)NRES * NH * 4);
  unsigned* SMAX  = (unsigned*)carve((size_t)NRES * NH * 4);
  float*    SDEN  = (float*)carve((size_t)NRES * NH * 4);
  float*    MEAN  = (float*)carve(HD * 4);
  float*    VAR   = (float*)carve(HD * 4);
  float*    X2    = (float*)carve((size_t)NRES * DH * 4);
  float*    POOL  = (float*)carve((size_t)NSSE * DH * 4);
  float*    SCNT  = (float*)carve((size_t)NSSE * 4);
  float*    XSSEF = (float*)carve((size_t)NSSE * SSEIN * 4);
  float*    XSSEO = (float*)carve((size_t)NSSE * DH * 4);
  float*    XDOM  = (float*)carve((size_t)NDOM * DH * 4);
  float*    DCNT  = (float*)carve((size_t)NDOM * 4);
  float*    DEG   = (float*)carve((size_t)NDOM * 4);
  float*    HDOM  = (float*)carve((size_t)NDOM * OUTC * 4);
  float*    ODOM  = (float*)carve((size_t)NDOM * OUTC * 4);
  float*    SSEG  = (float*)carve(DH * 4);
  float*    DMEAN = (float*)carve(OUTC * 4);
  float*    CAT   = (float*)carve(2 * OUTC * 4);

  // ===== Level 1, GAT #1 =====
  cvt_pad  <<<CDIV((long long)NRES * KP1, 256), 256, 0, stream>>>(x_res, ABF, NRES, FIN, KP1);
  cvt_pad_t<<<CDIV((long long)HD * KP1, 256), 256, 0, stream>>>(W1, WTBF, FIN, HD, KP1);
  wmma_gemm_tn<<<dim3(HD / (16 * NT), CDIV(NRES / 16, 8)), 256, 0, stream>>>(ABF, WTBF, HB, NRES, HD, KP1);

  attn_scores<<<CDIV(NRES * NH, 256), 256, 0, stream>>>(HB, a_src1, a_dst1, ES, ED, NRES);
  fill_u32<<<CDIV(NRES * NH, 256), 256, 0, stream>>>(SMAX, ENC_NEG_INF, (long long)NRES * NH);
  edge_max<<<CDIV((long long)(ERES + NRES) * NH, 256), 256, 0, stream>>>(src_r, dst_r, ERES, NRES, ES, ED, SMAX);
  fill_f32<<<CDIV(NRES * NH, 256), 256, 0, stream>>>(SDEN, 0.f, (long long)NRES * NH);
  edge_den<<<CDIV((long long)(ERES + NRES) * NH, 256), 256, 0, stream>>>(src_r, dst_r, ERES, NRES, ES, ED, SMAX, SDEN);
  fill_bias<<<CDIV((long long)NRES * HD, 256), 256, 0, stream>>>(OB, b1, NRES, HD);
  edge_acc<<<dim3((ERES + NRES) * NH), 128, 0, stream>>>(src_r, dst_r, ERES, NRES, ES, ED, SMAX, SDEN, HB, OB);

  bn_stats<<<HD, 256, 0, stream>>>(OB, NRES, HD, MEAN, VAR);
  bn_relu<<<CDIV((long long)NRES * HD, 256), 256, 0, stream>>>(OB, NRES, HD, MEAN, VAR, g1, be1);

  // ===== Level 1, GAT #2 (concat=False) =====
  cvt_pad  <<<CDIV((long long)NRES * HD, 256), 256, 0, stream>>>(OB, ABF, NRES, HD, HD);
  cvt_pad_t<<<CDIV(HD * HD, 256), 256, 0, stream>>>(W2, WTBF, HD, HD, HD);
  wmma_gemm_tn<<<dim3(HD / (16 * NT), CDIV(NRES / 16, 8)), 256, 0, stream>>>(ABF, WTBF, HB, NRES, HD, HD);

  attn_scores<<<CDIV(NRES * NH, 256), 256, 0, stream>>>(HB, a_src2, a_dst2, ES, ED, NRES);
  fill_u32<<<CDIV(NRES * NH, 256), 256, 0, stream>>>(SMAX, ENC_NEG_INF, (long long)NRES * NH);
  edge_max<<<CDIV((long long)(ERES + NRES) * NH, 256), 256, 0, stream>>>(src_r, dst_r, ERES, NRES, ES, ED, SMAX);
  fill_f32<<<CDIV(NRES * NH, 256), 256, 0, stream>>>(SDEN, 0.f, (long long)NRES * NH);
  edge_den<<<CDIV((long long)(ERES + NRES) * NH, 256), 256, 0, stream>>>(src_r, dst_r, ERES, NRES, ES, ED, SMAX, SDEN);
  fill_f32<<<CDIV((long long)NRES * HD, 256), 256, 0, stream>>>(OB, 0.f, (long long)NRES * HD);
  edge_acc<<<dim3((ERES + NRES) * NH), 128, 0, stream>>>(src_r, dst_r, ERES, NRES, ES, ED, SMAX, SDEN, HB, OB);
  head_mean<<<CDIV((long long)NRES * DH, 256), 256, 0, stream>>>(OB, b2, X2, NRES);

  bn_stats<<<DH, 256, 0, stream>>>(X2, NRES, DH, MEAN, VAR);
  bn_relu<<<CDIV((long long)NRES * DH, 256), 256, 0, stream>>>(X2, NRES, DH, MEAN, VAR, g2, be2);

  // ===== pool residues -> SSEs =====
  fill_f32<<<CDIV(NSSE * DH, 256), 256, 0, stream>>>(POOL, 0.f, (long long)NSSE * DH);
  fill_f32<<<CDIV(NSSE, 256), 256, 0, stream>>>(SCNT, 0.f, NSSE);
  pool_cnt<<<CDIV(NRES, 256), 256, 0, stream>>>(res_sid, SCNT, NRES);
  pool_add<<<CDIV((long long)NRES * DH, 256), 256, 0, stream>>>(X2, res_sid, POOL, NRES, DH);
  build_sse<<<CDIV(NSSE * SSEIN, 256), 256, 0, stream>>>(POOL, SCNT, ss_type, XSSEF);

  // ===== Level 2: SSE GAT (concat=False) =====
  cvt_pad  <<<CDIV(NSSE * KPS, 256), 256, 0, stream>>>(XSSEF, ABF, NSSE, SSEIN, KPS);
  cvt_pad_t<<<CDIV(HD * KPS, 256), 256, 0, stream>>>(Ws, WTBF, SSEIN, HD, KPS);
  wmma_gemm_tn<<<dim3(HD / (16 * NT), CDIV(NSSE / 16, 8)), 256, 0, stream>>>(ABF, WTBF, HB, NSSE, HD, KPS);

  attn_scores<<<CDIV(NSSE * NH, 256), 256, 0, stream>>>(HB, a_srcs, a_dsts, ES, ED, NSSE);
  fill_u32<<<CDIV(NSSE * NH, 256), 256, 0, stream>>>(SMAX, ENC_NEG_INF, (long long)NSSE * NH);
  edge_max<<<CDIV((long long)(ESSE + NSSE) * NH, 256), 256, 0, stream>>>(src_s, dst_s, ESSE, NSSE, ES, ED, SMAX);
  fill_f32<<<CDIV(NSSE * NH, 256), 256, 0, stream>>>(SDEN, 0.f, (long long)NSSE * NH);
  edge_den<<<CDIV((long long)(ESSE + NSSE) * NH, 256), 256, 0, stream>>>(src_s, dst_s, ESSE, NSSE, ES, ED, SMAX, SDEN);
  fill_f32<<<CDIV((long long)NSSE * HD, 256), 256, 0, stream>>>(OB, 0.f, (long long)NSSE * HD);
  edge_acc<<<dim3((ESSE + NSSE) * NH), 128, 0, stream>>>(src_s, dst_s, ESSE, NSSE, ES, ED, SMAX, SDEN, HB, OB);
  head_mean<<<CDIV(NSSE * DH, 256), 256, 0, stream>>>(OB, bs, XSSEO, NSSE);

  bn_stats<<<DH, 256, 0, stream>>>(XSSEO, NSSE, DH, MEAN, VAR);
  bn_relu<<<CDIV(NSSE * DH, 256), 256, 0, stream>>>(XSSEO, NSSE, DH, MEAN, VAR, gs, bes);

  // ===== pool SSEs -> domains =====
  fill_f32<<<CDIV(NDOM * DH, 256), 256, 0, stream>>>(XDOM, 0.f, NDOM * DH);
  fill_f32<<<1, 256, 0, stream>>>(DCNT, 0.f, NDOM);
  dom_cnt<<<CDIV(NSSE, 256), 256, 0, stream>>>(DCNT);
  dom_pool<<<CDIV(NSSE * DH, 256), 256, 0, stream>>>(XSSEO, XDOM);
  divide_rows<<<CDIV(NDOM * DH, 256), 256, 0, stream>>>(XDOM, DCNT, NDOM, DH);

  // ===== Level 3: domain GCN =====
  fill_f32<<<1, 256, 0, stream>>>(DEG, 0.f, NDOM);
  deg_count<<<1, 256, 0, stream>>>(src_d, dst_d, EDOM, NDOM, DEG);
  small_gemm<<<NDOM, OUTC, 0, stream>>>(XDOM, Wd, HDOM, DH, OUTC);
  fill_bias<<<CDIV(NDOM * OUTC, 256), 256, 0, stream>>>(ODOM, bd, NDOM, OUTC);
  gcn_acc<<<EDOM + NDOM, OUTC, 0, stream>>>(src_d, dst_d, EDOM, NDOM, DEG, HDOM, ODOM, OUTC);
  bn_stats<<<OUTC, 256, 0, stream>>>(ODOM, NDOM, OUTC, MEAN, VAR);
  bn_relu<<<CDIV(NDOM * OUTC, 256), 256, 0, stream>>>(ODOM, NDOM, OUTC, MEAN, VAR, gd, bed);

  // ===== readout =====
  colmean<<<DH, 256, 0, stream>>>(XSSEO, NSSE, DH, SSEG);
  colmean<<<OUTC, 256, 0, stream>>>(ODOM, NDOM, OUTC, DMEAN);
  proj_cat<<<1, OUTC, 0, stream>>>(SSEG, Wproj, bproj, DMEAN, CAT);
  final_out<<<1, OUTC, 0, stream>>>(CAT, Wread, bread, (float*)d_out);
}
